// BiCAF_89867895701884
// MI455X (gfx1250) — matrix operand order repaired and checked by reading, not compiled, not run on hardware
//
#include <hip/hip_runtime.h>

// ---------------------------------------------------------------------------
// BiCAF fused pipeline for gfx1250 (MI455X): dw3x3 + pw(1x1) via WMMA bf16,
// 8x8 block attention via WMMA bf16, fused residual + GroupNorm.
// wave32, 256-thread workgroups (8 waves). All matrix math on v_wmma_f32_16x16x32_bf16.
// ---------------------------------------------------------------------------

typedef __attribute__((ext_vector_type(16))) __bf16 v16bf;
typedef __attribute__((ext_vector_type(8)))  float  v8f;

#define NBLK 2048          // blocks per source image set: 8 batch * 16 * 16
#define EPS_ 1e-5f

union ABu { v16bf v; uint4 q[2]; __bf16 h[16]; };

__device__ __forceinline__ v8f wmma_bf16(v16bf a, v16bf b, v8f c) {
  return __builtin_amdgcn_wmma_f32_16x16x32_bf16(false, a, false, b, (short)0, c,
                                                 false, false);
}

// A-matrix fragment (16x32 bf16): lane<16 -> row M=lane, K=0..15 contiguous;
// lane>=16 -> row M=lane-16, K=16..31. Caller passes full row index.
__device__ __forceinline__ v16bf load_frag_a(const __bf16* base, int stride,
                                             int row, int kofs, int lane) {
  int kb = kofs + ((lane & 16) ? 16 : 0);
  const __bf16* p = base + (size_t)row * stride + kb;
  ABu u;
  u.q[0] = *(const uint4*)(p);
  u.q[1] = *(const uint4*)(p + 8);
  return u.v;
}

// B-matrix fragment (32x16 bf16): lane<16 -> col N=lane, K=0..15 contiguous;
// lane>=16 -> col N=lane-16, K=16..31. Caller passes full column index.
__device__ __forceinline__ v16bf load_frag_b(const __bf16* base, int stride,
                                             int col, int kofs, int lane) {
  int kb = kofs + ((lane & 16) ? 16 : 0);
  const __bf16* p = base + (size_t)col * stride + kb;
  ABu u;
  u.q[0] = *(const uint4*)(p);
  u.q[1] = *(const uint4*)(p + 8);
  return u.v;
}

// Pre-packed B fragment: 32 lanes x 16 bf16 contiguous per (n_tile, k_tile).
__device__ __forceinline__ v16bf load_frag_bpk(const __bf16* pk, int nt, int kt,
                                               int lane) {
  const __bf16* p = pk + ((size_t)((nt * 8 + kt) * 32) + lane) * 16;
  ABu u;
  u.q[0] = *(const uint4*)(p);
  u.q[1] = *(const uint4*)(p + 8);
  return u.v;
}

// ---------------------------------------------------------------------------
// Kernel 0: pack pointwise weights (O x 256 f32) into WMMA B-fragment order.
// pk[nt][kt][lane][16] = W[nt*16 + (lane&15)][kt*32 + (lane<16?0:16) + j]
// ---------------------------------------------------------------------------
__device__ __forceinline__ void pack_one(const float* w, __bf16* pk, int e) {
  int nt   = e >> 12;          // 8*32*16 elements per n-tile
  int rem  = e & 4095;
  int kt   = rem >> 9;         // 32*16 per k-tile
  int lane = (rem >> 4) & 31;
  int j    = rem & 15;
  int o = nt * 16 + (lane & 15);
  int c = kt * 32 + ((lane & 16) ? 16 : 0) + j;
  pk[e] = (__bf16)w[o * 256 + c];
}

__global__ __launch_bounds__(256) void pack_wmma_weights(
    const float* __restrict__ qpw, const float* __restrict__ kpw,
    const float* __restrict__ vpw, __bf16* __restrict__ qpk,
    __bf16* __restrict__ kpk, __bf16* __restrict__ vpk) {
  int idx = blockIdx.x * 256 + threadIdx.x;   // 288*256 = 73728 total
  if (idx < 4096)            pack_one(qpw, qpk, idx);
  else if (idx < 8192)       pack_one(kpw, kpk, idx - 4096);
  else if (idx < 8192+65536) pack_one(vpw, vpk, idx - 8192);
}

// ---------------------------------------------------------------------------
// Kernel 1: fused dw3x3 + pointwise (WMMA) producing q/k (16ch, zero-padded to
// 32) and v (256ch) in attention-block layout, bf16.
//   qbuf/kbuf: [src][blk][64 pix][32 ch]   (A for QK^T, B for K)
//   vbuf:      [src][blk][256 ch][64 pix]  (A for AV gemm)
// One workgroup = one (src, batch, 8x8 block). 8 k-chunks of 32 channels.
// ---------------------------------------------------------------------------
__global__ __launch_bounds__(256) void qkv_kernel(
    const float* __restrict__ img, const float* __restrict__ mask,
    const float* __restrict__ qdw_w, const float* __restrict__ qdw_b,
    const float* __restrict__ kdw_w, const float* __restrict__ kdw_b,
    const float* __restrict__ vdw_w, const float* __restrict__ vdw_b,
    const __bf16* __restrict__ qpk, const __bf16* __restrict__ kpk,
    const __bf16* __restrict__ vpk,
    const float* __restrict__ qpw_b, const float* __restrict__ kpw_b,
    const float* __restrict__ vpw_b,
    __bf16* __restrict__ qbuf, __bf16* __restrict__ kbuf,
    __bf16* __restrict__ vbuf) {
  const int tid  = threadIdx.x;
  const int lane = tid & 31;
  const int wave = tid >> 5;

  const int g   = blockIdx.x;
  const int src = g >> 11;
  const int blk = g & 2047;
  const int b   = blk >> 8;
  const int by  = (blk >> 4) & 15;
  const int bx  = blk & 15;
  const float* x = src ? mask : img;

  __shared__ float  sh_in[32][10][12];   // halo tile, 32-ch chunk
  __shared__ float  sh_w[3][32][9];      // dw taps for q/k/v
  __shared__ float  sh_b[3][32];         // dw biases
  __shared__ __bf16 sh_a[3][64 * 40];    // dw outputs, A-matrix staging (pad 40)

  v8f zero = {0.f, 0.f, 0.f, 0.f, 0.f, 0.f, 0.f, 0.f};
  v8f accv[4][2];
#pragma unroll
  for (int m = 0; m < 4; ++m)
#pragma unroll
    for (int j = 0; j < 2; ++j) accv[m][j] = zero;
  v8f accqk = zero;

  const int y0 = by * 8 - 1;
  const int x0 = bx * 8 - 1;

  for (int kc = 0; kc < 8; ++kc) {
    __syncthreads();
    // ---- stage halo tile (32 ch x 10 x 10) + dw weights/biases ----
    for (int idx = tid; idx < 32 * 100; idx += 256) {
      int ch  = idx / 100;
      int rem = idx - ch * 100;
      int iy  = rem / 10, ix = rem - iy * 10;
      int yy = y0 + iy, xx = x0 + ix;
      float v = 0.f;
      if ((unsigned)yy < 128u && (unsigned)xx < 128u)
        v = x[(((size_t)b * 256 + (kc * 32 + ch)) * 128 + yy) * 128 + xx];
      sh_in[ch][iy][ix] = v;
    }
    for (int idx = tid; idx < 3 * 32 * 9; idx += 256) {
      int f = idx / 288, rem = idx - f * 288;
      int c = rem / 9, t = rem - c * 9;
      const float* wsrc = (f == 0) ? qdw_w : (f == 1) ? kdw_w : vdw_w;
      sh_w[f][c][t] = wsrc[(kc * 32 + c) * 9 + t];
    }
    if (tid < 96) {
      int f = tid >> 5, c = tid & 31;
      const float* bsrc = (f == 0) ? qdw_b : (f == 1) ? kdw_b : vdw_b;
      sh_b[f][c] = bsrc[kc * 32 + c];
    }
    __syncthreads();

    // ---- depthwise 3x3 -> bf16 A tiles (64 pix x 32 K) for 3 filters ----
    for (int idx = tid; idx < 3 * 64 * 32; idx += 256) {
      int f   = idx >> 11;
      int rem = idx & 2047;
      int ch  = rem >> 6;
      int p   = rem & 63;
      int py = p >> 3, px = p & 7;
      float acc = sh_b[f][ch];
#pragma unroll
      for (int ky = 0; ky < 3; ++ky)
#pragma unroll
        for (int kx = 0; kx < 3; ++kx)
          acc += sh_w[f][ch][ky * 3 + kx] * sh_in[ch][py + ky][px + kx];
      sh_a[f][p * 40 + ch] = (__bf16)acc;
    }
    __syncthreads();

    // ---- WMMA accumulate: v (all waves), q (waves 0-3), k (waves 4-7) ----
    v16bf av[4];
#pragma unroll
    for (int m = 0; m < 4; ++m)
      av[m] = load_frag_a(&sh_a[2][0], 40, m * 16 + (lane & 15), 0, lane);
#pragma unroll
    for (int j = 0; j < 2; ++j) {
      v16bf bv = load_frag_bpk(vpk, wave * 2 + j, kc, lane);
#pragma unroll
      for (int m = 0; m < 4; ++m) accv[m][j] = wmma_bf16(av[m], bv, accv[m][j]);
    }
    int f = (wave < 4) ? 0 : 1;
    int mt = wave & 3;
    v16bf af = load_frag_a(&sh_a[f][0], 40, mt * 16 + (lane & 15), 0, lane);
    v16bf bf_ = load_frag_bpk(f ? kpk : qpk, 0, kc, lane);
    accqk = wmma_bf16(af, bf_, accqk);
  }
  __syncthreads();

  // ---- stores: block-layout bf16 ----
  const size_t blkbase = (size_t)(src * NBLK + blk);
  // v: [blk][ch][pix]; D: M=pixel row = m*16 + r (+8 upper lanes), N=channel
#pragma unroll
  for (int m = 0; m < 4; ++m)
#pragma unroll
    for (int j = 0; j < 2; ++j) {
      int n = (wave * 2 + j) * 16 + (lane & 15);
      float bias = vpw_b[n];
      int p0 = m * 16 + ((lane & 16) ? 8 : 0);
      alignas(16) __bf16 t8[8];
#pragma unroll
      for (int r = 0; r < 8; ++r) t8[r] = (__bf16)(accv[m][j][r] + bias);
      *(uint4*)(vbuf + (blkbase * 256 + n) * 64 + p0) = *(const uint4*)t8;
    }
  // q/k: [blk][pix][32ch], channels 16..31 zeroed for K=32 padding
  {
    int f = (wave < 4) ? 0 : 1;
    int mt = wave & 3;
    __bf16* dst = f ? kbuf : qbuf;
    const float* pb = f ? kpw_b : qpw_b;
    int n = lane & 15;
    float bias = pb[n];
#pragma unroll
    for (int r = 0; r < 8; ++r) {
      int p = mt * 16 + r + ((lane & 16) ? 8 : 0);
      dst[(blkbase * 64 + p) * 32 + n] = (__bf16)(accqk[r] + bias);
    }
  }
  for (int idx = tid; idx < 64 * 16; idx += 256) {
    int p = idx >> 4, c2 = 16 + (idx & 15);
    qbuf[(blkbase * 64 + p) * 32 + c2] = (__bf16)0.f;
    kbuf[(blkbase * 64 + p) * 32 + c2] = (__bf16)0.f;
  }
}

// ---------------------------------------------------------------------------
// Kernel 2: block attention. One workgroup = one (pass, block).
//   S = Q^T K / sqrt(16)  (64x64, K=32 zero-padded), softmax rows,
//   out = V * P^T  -> enh [pass][blk][64 pix][256 ch] bf16.
// pass 0: q=mask, k/v=img (mask_enhanced). pass 1: q=img, k/v=mask.
// ---------------------------------------------------------------------------
__global__ __launch_bounds__(256) void attn_kernel(
    const __bf16* __restrict__ qbuf, const __bf16* __restrict__ kbuf,
    const __bf16* __restrict__ vbuf, __bf16* __restrict__ enh) {
  const int tid  = threadIdx.x;
  const int lane = tid & 31;
  const int wave = tid >> 5;

  const int g    = blockIdx.x;
  const int pass = g >> 11;
  const int blk  = g & 2047;
  const int qsrc  = pass ? 0 : 1;
  const int kvsrc = 1 - qsrc;

  __shared__ float  sh_S[64][66];
  __shared__ __bf16 sh_P[64][72];

  const __bf16* qb = qbuf + (size_t)(qsrc  * NBLK + blk) * 64 * 32;
  const __bf16* kb = kbuf + (size_t)(kvsrc * NBLK + blk) * 64 * 32;
  const __bf16* vb = vbuf + (size_t)(kvsrc * NBLK + blk) * 256 * 64;

  v8f zero = {0.f, 0.f, 0.f, 0.f, 0.f, 0.f, 0.f, 0.f};

  // ---- phase A: S tiles (4x4 of 16x16), 2 per wave ----
#pragma unroll
  for (int tt = 0; tt < 2; ++tt) {
    int t  = wave * 2 + tt;
    int mi = t >> 2, ni = t & 3;
    v16bf a = load_frag_a(qb, 32, mi * 16 + (lane & 15), 0, lane);
    v16bf bq = load_frag_b(kb, 32, ni * 16 + (lane & 15), 0, lane);
    v8f acc = wmma_bf16(a, bq, zero);
    int row0 = mi * 16 + ((lane & 16) ? 8 : 0);
    int col  = ni * 16 + (lane & 15);
#pragma unroll
    for (int r = 0; r < 8; ++r) sh_S[row0 + r][col] = acc[r] * 0.25f;
  }
  __syncthreads();

  // ---- softmax rows (f32) -> P bf16 ----
  if (tid < 64) {
    float m = -1e30f;
    for (int q = 0; q < 64; ++q) m = fmaxf(m, sh_S[tid][q]);
    float s = 0.f;
    for (int q = 0; q < 64; ++q) {
      float e = __expf(sh_S[tid][q] - m);
      sh_S[tid][q] = e;
      s += e;
    }
    float inv = 1.f / s;
    for (int q = 0; q < 64; ++q) sh_P[tid][q] = (__bf16)(sh_S[tid][q] * inv);
  }
  __syncthreads();

  // ---- phase B: out = V (256x64) * P^T (64x64); 8 tiles per wave ----
  v8f acc2[2][4];
#pragma unroll
  for (int ci = 0; ci < 2; ++ci)
#pragma unroll
    for (int ni = 0; ni < 4; ++ni) acc2[ci][ni] = zero;

#pragma unroll
  for (int kk = 0; kk < 2; ++kk) {
    int kc2 = kk * 32;
    v16bf bP[4];
#pragma unroll
    for (int ni = 0; ni < 4; ++ni)
      bP[ni] = load_frag_b(&sh_P[0][0], 72, ni * 16 + (lane & 15), kc2, lane);
#pragma unroll
    for (int ci = 0; ci < 2; ++ci) {
      int ct = wave * 2 + ci;
      v16bf aV = load_frag_a(vb, 64, ct * 16 + (lane & 15), kc2, lane);
#pragma unroll
      for (int ni = 0; ni < 4; ++ni)
        acc2[ci][ni] = wmma_bf16(aV, bP[ni], acc2[ci][ni]);
    }
  }

  // store: enh[pass][blk][p][c]; per lane p fixed, 8 consecutive c -> b128
  __bf16* eb = enh + (size_t)(pass * NBLK + blk) * 64 * 256;
#pragma unroll
  for (int ci = 0; ci < 2; ++ci) {
    int c0 = (wave * 2 + ci) * 16 + ((lane & 16) ? 8 : 0);
#pragma unroll
    for (int ni = 0; ni < 4; ++ni) {
      int p = ni * 16 + (lane & 15);
      alignas(16) __bf16 t8[8];
#pragma unroll
      for (int r = 0; r < 8; ++r) t8[r] = (__bf16)acc2[ci][ni][r];
      *(uint4*)(eb + (size_t)p * 256 + c0) = *(const uint4*)t8;
    }
  }
}

// ---------------------------------------------------------------------------
// Kernel 3: GroupNorm stats over fused = a*enh0 + b*enh1 + img + mask.
// One workgroup per (batch, group): 8 ch x 16384 px = 131072 elements.
// ---------------------------------------------------------------------------
__global__ __launch_bounds__(256) void stats_kernel(
    const float* __restrict__ img, const float* __restrict__ mask,
    const __bf16* __restrict__ enh, const float* __restrict__ alpha,
    const float* __restrict__ beta, float* __restrict__ stats) {
  const int tid = threadIdx.x;
  const int bg = blockIdx.x;
  const int b = bg >> 5;
  const int c0 = (bg & 31) * 8;
  const float a = alpha[0], be = beta[0];
  const size_t E1 = (size_t)NBLK * 64 * 256;

  float s = 0.f, ss = 0.f;
  for (int idx = tid; idx < 8 * 16384; idx += 256) {
    int co = idx >> 14;
    int pix = idx & 16383;
    int h = pix >> 7, w = pix & 127;
    int c = c0 + co;
    size_t ni = ((size_t)b * 256 + c) * 16384 + pix;
    int n = b * 256 + ((h >> 3) << 4) + (w >> 3);
    int p = ((h & 7) << 3) + (w & 7);
    size_t eo = ((size_t)n * 64 + p) * 256 + c;
    float f = a * (float)enh[eo] + be * (float)enh[E1 + eo] + img[ni] + mask[ni];
    s += f;
    ss += f * f;
  }
  __shared__ float rs[256], rq[256];
  rs[tid] = s; rq[tid] = ss;
  __syncthreads();
  for (int o = 128; o > 0; o >>= 1) {
    if (tid < o) { rs[tid] += rs[tid + o]; rq[tid] += rq[tid + o]; }
    __syncthreads();
  }
  if (tid == 0) {
    float mean = rs[0] * (1.f / 131072.f);
    float var = rq[0] * (1.f / 131072.f) - mean * mean;
    stats[bg * 2] = mean;
    stats[bg * 2 + 1] = rsqrtf(var + EPS_);
  }
}

// ---------------------------------------------------------------------------
// Kernel 4: fuse + normalize, NCHW f32 output.
// ---------------------------------------------------------------------------
__global__ __launch_bounds__(256) void finalize_kernel(
    const float* __restrict__ img, const float* __restrict__ mask,
    const __bf16* __restrict__ enh, const float* __restrict__ alpha,
    const float* __restrict__ beta, const float* __restrict__ stats,
    const float* __restrict__ gnw, const float* __restrict__ gnb,
    float* __restrict__ out) {
  int idx = blockIdx.x * 256 + threadIdx.x;   // 2^25 total
  int b = idx >> 22;
  int c = (idx >> 14) & 255;
  int pix = idx & 16383;
  int h = pix >> 7, w = pix & 127;
  const float a = alpha[0], be = beta[0];
  const size_t E1 = (size_t)NBLK * 64 * 256;

  int n = b * 256 + ((h >> 3) << 4) + (w >> 3);
  int p = ((h & 7) << 3) + (w & 7);
  size_t eo = ((size_t)n * 64 + p) * 256 + c;
  float f = a * (float)enh[eo] + be * (float)enh[E1 + eo] + img[idx] + mask[idx];
  int g = c >> 3;
  float mean = stats[(b * 32 + g) * 2];
  float rstd = stats[(b * 32 + g) * 2 + 1];
  out[idx] = (f - mean) * rstd * gnw[c] + gnb[c];
}

// ---------------------------------------------------------------------------
extern "C" void kernel_launch(void* const* d_in, const int* in_sizes, int n_in,
                              void* d_out, int out_size, void* d_ws,
                              size_t ws_size, hipStream_t stream) {
  const float* img   = (const float*)d_in[0];
  const float* mask  = (const float*)d_in[1];
  const float* qdw_w = (const float*)d_in[2];
  const float* qdw_b = (const float*)d_in[3];
  const float* kdw_w = (const float*)d_in[4];
  const float* kdw_b = (const float*)d_in[5];
  const float* vdw_w = (const float*)d_in[6];
  const float* vdw_b = (const float*)d_in[7];
  const float* qpw_w = (const float*)d_in[8];
  const float* qpw_b = (const float*)d_in[9];
  const float* kpw_w = (const float*)d_in[10];
  const float* kpw_b = (const float*)d_in[11];
  const float* vpw_w = (const float*)d_in[12];
  const float* vpw_b = (const float*)d_in[13];
  const float* alpha = (const float*)d_in[14];
  const float* beta  = (const float*)d_in[15];
  const float* gnw   = (const float*)d_in[16];
  const float* gnb   = (const float*)d_in[17];

  char* ws = (char*)d_ws;
  __bf16* qpk   = (__bf16*)(ws);              //   8 KB
  __bf16* kpk   = (__bf16*)(ws + 8192);       //   8 KB
  __bf16* vpk   = (__bf16*)(ws + 16384);      // 128 KB
  float*  stats = (float*)(ws + 147456);      //   2 KB
  __bf16* qbuf  = (__bf16*)(ws + (1 << 18));  //  16 MB: [2][2048][64][32]
  __bf16* kbuf  = qbuf + (size_t)2 * NBLK * 64 * 32;   // 16 MB
  __bf16* vbuf  = kbuf + (size_t)2 * NBLK * 64 * 32;   // 128 MB: [2][2048][256][64]
  __bf16* enh   = vbuf + (size_t)2 * NBLK * 256 * 64;  // 128 MB: [2][2048][64][256]

  pack_wmma_weights<<<288, 256, 0, stream>>>(qpw_w, kpw_w, vpw_w, qpk, kpk, vpk);
  qkv_kernel<<<2 * NBLK, 256, 0, stream>>>(img, mask, qdw_w, qdw_b, kdw_w, kdw_b,
                                           vdw_w, vdw_b, qpk, kpk, vpk, qpw_b,
                                           kpw_b, vpw_b, qbuf, kbuf, vbuf);
  attn_kernel<<<2 * NBLK, 256, 0, stream>>>(qbuf, kbuf, vbuf, enh);
  stats_kernel<<<256, 256, 0, stream>>>(img, mask, enh, alpha, beta, stats);
  finalize_kernel<<<131072, 256, 0, stream>>>(img, mask, enh, alpha, beta, stats,
                                              gnw, gnb, (float*)d_out);
}
